// UnidirecLSTMLayerWithDropoutBefore_8761733284420
// MI455X (gfx1250) — matrix operand order repaired and checked by reading, not compiled, not run on hardware
//
#include <hip/hip_runtime.h>
#include <hip/hip_bf16.h>
#include <stdint.h>

// ---------------------------------------------------------------------------
// UnidirecLSTMLayerWithDropoutBefore  (B=64, T=2048, D=256, H=256)
//  Phase 0: pack Wi/Wh -> bf16, transposed [4H][K] for contiguous B-fragments
//  Phase 1: Z = dropout(x) @ Wi + b, stored in WMMA C-fragment layout.
//           Input tile staged into LDS by the Tensor Data Mover (TDM).
//  Phase 2: persistent 1-workgroup recurrence, h in LDS (bf16, double buffer),
//           c in VGPRs, Wh streamed from L2 once per (kc,gate) per step.
// ---------------------------------------------------------------------------

#define B_  64
#define T_  2048
#define D_  256
#define H_  256
#define G4H 1024                 // 4*H
#define KC  8                    // K chunks of 32 (K=256)
#define DROP_SCALE (1.0f / 0.9f)

typedef __bf16 bf16x8  __attribute__((ext_vector_type(8)));
typedef __bf16 bf16x16 __attribute__((ext_vector_type(16)));
typedef float  v8f     __attribute__((ext_vector_type(8)));
typedef uint32_t u32x4 __attribute__((ext_vector_type(4)));
typedef uint32_t u32x8 __attribute__((ext_vector_type(8)));

union FragU { bf16x16 v; bf16x8 h[2]; };

__device__ __forceinline__ uint32_t wang_hash(uint32_t s) {
  s = (s ^ 61u) ^ (s >> 16); s *= 9u; s ^= s >> 4;
  s *= 0x27d4eb2du; s ^= s >> 15; return s;
}

__device__ __forceinline__ float sigmoidf_(float x) {
  return 1.0f / (1.0f + __expf(-x));
}

// A fragment (16x32 bf16) from an LDS tile row-major [rows][256]
__device__ __forceinline__ bf16x16 load_a_frag(const __bf16* lds_base,
                                               int row, int kc, int lane) {
  const int kb = kc * 32 + ((lane >= 16) ? 8 : 0);
  FragU u;
  u.h[0] = *(const bf16x8*)(lds_base + (size_t)row * 256 + kb);
  u.h[1] = *(const bf16x8*)(lds_base + (size_t)row * 256 + kb + 16);
  return u.v;
}

// B fragment (32x16 bf16) from packed transposed weights WT[col][k] (k stride 1)
__device__ __forceinline__ bf16x16 load_b_frag(const __bf16* WT,
                                               int col, int kc, int lane) {
  const int kb = kc * 32 + ((lane >= 16) ? 8 : 0);
  const __bf16* p = WT + (size_t)col * 256 + kb;
  FragU u;
  u.h[0] = *(const bf16x8*)(p);
  u.h[1] = *(const bf16x8*)(p + 16);
  return u.v;
}

// C-fragment index in Z: tile (t, rt, ct), per-lane 8 contiguous floats
__device__ __forceinline__ size_t z_frag_off(int t, int rt, int ct, int lane) {
  return ((((size_t)t * 4 + rt) * 64 + ct) * 32 + lane) * 8;
}

// ---------------------------------------------------------------------------
// Phase 0: convert + transpose Wi [256][1024], Wh [256][1024] -> bf16 [1024][256]
// ---------------------------------------------------------------------------
__global__ void pack_weights_kernel(const float* __restrict__ Wi,
                                    const float* __restrict__ Wh,
                                    __bf16* __restrict__ WiT,
                                    __bf16* __restrict__ WhT) {
  int idx = blockIdx.x * 256 + threadIdx.x;       // [0, 1024*256)
  int col = idx >> 8;                             // [0,1024)
  int k   = idx & 255;                            // [0,256)
  WiT[(size_t)col * 256 + k] = (__bf16)Wi[(size_t)k * G4H + col];
  WhT[(size_t)col * 256 + k] = (__bf16)Wh[(size_t)k * G4H + col];
}

// ---------------------------------------------------------------------------
// Phase 1: Z = dropout(x) @ Wi + bias, stored in C-fragment layout.
// Block = 256 threads (8 waves). Block -> (t = blk>>2, rt = blk&3):
// rows b in [16*rt, 16*rt+16) at fixed t. Wave w -> col tiles 8w..8w+8.
// The 16x256 fp32 input tile (rows strided by T*D) is staged into LDS by one
// TDM tensor_load_to_lds (2D tile descriptor), tracked with TENSORcnt.
// ---------------------------------------------------------------------------
__global__ __launch_bounds__(256, 2)
void input_proj_kernel(const float* __restrict__ x,
                       const __bf16* __restrict__ WiT,
                       const float* __restrict__ bias,
                       float* __restrict__ Z) {
  __shared__ float  xraw[16][256];                // 16 KB TDM landing buffer
  __shared__ __bf16 xs[16][256];                  // 8 KB, dropout applied, bf16

  const int tid  = threadIdx.x;
  const int wave = tid >> 5;
  const int lane = tid & 31;
  const int t    = blockIdx.x >> 2;
  const int rt   = blockIdx.x & 3;

  // --- TDM: stage x[b0..b0+16)[t][0..256) -> xraw, one descriptor, wave 0 ---
  if (tid < 32) {
    const uint64_t ga  = (uint64_t)(uintptr_t)(x + ((size_t)(rt * 16) * T_ + t) * D_);
    const uint32_t lds = (uint32_t)(uintptr_t)&xraw[0][0];   // LDS byte offset

    u32x4 g0;
    g0[0] = 1u;                                   // count=1, user descriptor
    g0[1] = lds;                                  // lds_addr
    g0[2] = (uint32_t)ga;                         // global_addr[31:0]
    g0[3] = ((uint32_t)(ga >> 32) & 0x01FFFFFFu)  // global_addr[56:32]
            | (2u << 30);                         // type=2 ("image")

    u32x8 g1;
    g1[0] = (2u << 16);                           // data_size=4B, no multicast
    g1[1] = (256u << 16);                         // tensor_dim0[15:0]=256
    g1[2] = (16u << 16);                          // dim0 hi=0 | tensor_dim1=16
    g1[3] = (256u << 16);                         // dim1 hi=0 | tile_dim0=256
    g1[4] = 16u;                                  // tile_dim1=16, tile_dim2=0
    g1[5] = (uint32_t)(T_ * D_);                  // tensor_dim0_stride = 524288
    g1[6] = 0u;                                   // stride hi | dim1_stride lo
    g1[7] = 0u;

    u32x4 g2 = {0u, 0u, 0u, 0u};
    u32x4 g3 = {0u, 0u, 0u, 0u};

    asm volatile("tensor_load_to_lds %0, %1, %2, %3"
                 :
                 : "s"(g0), "s"(g1), "s"(g2), "s"(g3)
                 : "memory");
    __builtin_amdgcn_s_wait_tensorcnt(0);
  }
  __syncthreads();

  // dropout + fp32 -> bf16 out of the TDM-written staging buffer
  for (int e = tid; e < 16 * 256; e += 256) {
    int row = e >> 8, col = e & 255;              // row = b - 16*rt
    int b = rt * 16 + row;
    size_t gi = ((size_t)b * T_ + t) * 256 + col; // flat x index (< 2^25)
    float v = *(volatile const float*)&xraw[row][col];
    uint32_t h = wang_hash((uint32_t)gi);
    float u = (float)(h >> 8) * (1.0f / 16777216.0f);
    xs[row][col] = (__bf16)((u < 0.9f) ? v * DROP_SCALE : 0.0f);
  }
  __syncthreads();

  // each wave: 8 consecutive 16-col tiles, bias folded into accumulator
  v8f acc[8];
#pragma unroll
  for (int i = 0; i < 8; ++i) {
    float bv = bias[(wave * 8 + i) * 16 + (lane & 15)];
#pragma unroll
    for (int r = 0; r < 8; ++r) acc[i][r] = bv;
  }

#pragma unroll
  for (int kc = 0; kc < KC; ++kc) {
    bf16x16 a = load_a_frag(&xs[0][0], lane & 15, kc, lane);
#pragma unroll
    for (int i = 0; i < 8; ++i) {
      bf16x16 bfr = load_b_frag(WiT, (wave * 8 + i) * 16 + (lane & 15), kc, lane);
      acc[i] = __builtin_amdgcn_wmma_f32_16x16x32_bf16(
          false, a, false, bfr, (short)0, acc[i], false, false);
    }
  }

  // wide fragment-layout stores (2x b128 per tile)
#pragma unroll
  for (int i = 0; i < 8; ++i) {
    *(v8f*)(Z + z_frag_off(t, rt, wave * 8 + i, lane)) = acc[i];
  }
}

// ---------------------------------------------------------------------------
// Phase 2: persistent recurrence. 1 block, 512 threads (16 waves).
// Wave j owns h-columns [16j,16j+16) for all 4 gates (ct = g*16 + j).
// kc-outer loop: one B-fragment set serves all 4 row tiles (4x Wh reuse).
// ---------------------------------------------------------------------------
__global__ __launch_bounds__(512, 1)
void lstm_recurrence_kernel(const float* __restrict__ Z,
                            const __bf16* __restrict__ WhT,
                            const int* __restrict__ lens,
                            float* __restrict__ out) {
  __shared__ __bf16 hbuf[2][B_][H_];              // 64 KB double buffer
  __shared__ int    slens[B_];

  const int tid  = threadIdx.x;
  const int wave = tid >> 5;                      // 0..15 -> h-tile
  const int lane = tid & 31;
  const int hcol = wave * 16 + (lane & 15);

  for (int e = tid; e < B_ * H_; e += 512) {
    hbuf[0][e >> 8][e & 255] = (__bf16)0.0f;
  }
  if (tid < B_) slens[tid] = lens[tid];
  __syncthreads();

  float creg[4][8];
#pragma unroll
  for (int rt = 0; rt < 4; ++rt)
#pragma unroll
    for (int r = 0; r < 8; ++r) creg[rt][r] = 0.0f;

  float* c_last  = out;                           // [64][256]
  float* h_last  = out + 16384;                   // [64][256]
  float* outputs = out + 32768;                   // [64][2048][256]

  for (int t = 0; t < T_; ++t) {
    const int cur = t & 1, nxt = cur ^ 1;

    // speculative prefetch of next step's Z slice (256 KB / 512 threads)
    if (t + 1 < T_) {
      const char* pf = (const char*)(Z + (size_t)(t + 1) * B_ * G4H)
                       + (size_t)tid * 512;
      __builtin_prefetch(pf, 0, 3);
    }

    // all 16 accumulators start from precomputed x@Wi + b (fragment layout)
    v8f acc[4][4];                                // [rt][gate]
#pragma unroll
    for (int rt = 0; rt < 4; ++rt)
#pragma unroll
      for (int g = 0; g < 4; ++g)
        acc[rt][g] = *(const v8f*)(Z + z_frag_off(t, rt, g * 16 + wave, lane));

    // z += h_prev @ Wh ; B fragments loaded once per kc, reused over 4 rt
#pragma unroll
    for (int kc = 0; kc < KC; ++kc) {
      bf16x16 bfr[4];
#pragma unroll
      for (int g = 0; g < 4; ++g)
        bfr[g] = load_b_frag(WhT, g * 256 + hcol, kc, lane);
#pragma unroll
      for (int rt = 0; rt < 4; ++rt) {
        bf16x16 a = load_a_frag(&hbuf[cur][0][0], rt * 16 + (lane & 15), kc, lane);
#pragma unroll
        for (int g = 0; g < 4; ++g) {
          acc[rt][g] = __builtin_amdgcn_wmma_f32_16x16x32_bf16(
              false, a, false, bfr[g], (short)0, acc[rt][g], false, false);
        }
      }
    }

    // gates: i=acc[rt][0], f=acc[rt][1], g=acc[rt][2], o=acc[rt][3]
#pragma unroll
    for (int rt = 0; rt < 4; ++rt) {
#pragma unroll
      for (int r = 0; r < 8; ++r) {
        int row = rt * 16 + r + ((lane >> 4) << 3);
        float ig = sigmoidf_(acc[rt][0][r]);
        float fg = sigmoidf_(acc[rt][1][r]);
        float gg = tanhf(acc[rt][2][r]);
        float og = sigmoidf_(acc[rt][3][r]);
        float cn = fg * creg[rt][r] + ig * gg;
        float hn = og * tanhf(cn);
        creg[rt][r] = cn;
        hbuf[nxt][row][hcol] = (__bf16)hn;
        outputs[((size_t)row * T_ + t) * H_ + hcol] = hn;
        if (slens[row] - 1 == t) {
          c_last[row * H_ + hcol] = cn;
          h_last[row * H_ + hcol] = hn;
        }
      }
    }
    __syncthreads();
  }
}

// ---------------------------------------------------------------------------
extern "C" void kernel_launch(void* const* d_in, const int* in_sizes, int n_in,
                              void* d_out, int out_size, void* d_ws, size_t ws_size,
                              hipStream_t stream) {
  (void)in_sizes; (void)n_in; (void)out_size; (void)ws_size;

  const float* x    = (const float*)d_in[0];      // [64][2048][256]
  const float* Wi   = (const float*)d_in[1];      // [256][1024]
  const float* Wh   = (const float*)d_in[2];      // [256][1024]
  const float* bias = (const float*)d_in[3];      // [1024]
  const int*   lens = (const int*)d_in[4];        // [64]
  float* out = (float*)d_out;

  // workspace: WiT(512K) | WhT(512K) | Z fp32 fragment layout (512 MB)
  char* ws = (char*)d_ws;
  __bf16* WiT = (__bf16*)(ws);
  __bf16* WhT = (__bf16*)(ws + (size_t)512 * 1024);
  float*  Zbf = (float*) (ws + (size_t)1024 * 1024);

  pack_weights_kernel<<<1024, 256, 0, stream>>>(Wi, Wh, WiT, WhT);
  input_proj_kernel<<<(B_ * T_) / 16, 256, 0, stream>>>(x, WiT, bias, Zbf);
  lstm_recurrence_kernel<<<1, 512, 0, stream>>>(Zbf, WhT, lens, out);
}
